// HeteroGraphAutoencoder_5729486373619
// MI455X (gfx1250) — compile-verified, matrix operand order checked by reading, not run on hardware
//
#include <hip/hip_runtime.h>
#include <hip/hip_bf16.h>
#include <math.h>

#define N_INT 50000
#define N_EXT 20000
#define E_II 800000
#define E_IE 300000
#define E_EI 300000

typedef __attribute__((ext_vector_type(16))) __bf16 v16bf;
typedef __attribute__((ext_vector_type(8)))  float  v8f;

// ---------- helpers ----------
__device__ __forceinline__ unsigned short f2bf(float f) {
  unsigned int u = __float_as_uint(f);
  u += 0x7FFFu + ((u >> 16) & 1u);          // round-to-nearest-even
  return (unsigned short)(u >> 16);
}
__device__ __forceinline__ float bf2f(unsigned int bits16) {
  return __uint_as_float(bits16 << 16);
}
__device__ __forceinline__ void atomicMaxF(float* addr, float v) {
  // monotone float<->int bit mapping trick; m[] is initialized to -inf
  if (v >= 0.f) atomicMax((int*)addr, __float_as_int(v));
  else          atomicMin((unsigned int*)addr, __float_as_uint(v));
}

union FragBF { uint4 u[2]; v16bf v; };

// ---------- WMMA GEMM: C[N,M](bf16) = A[N,K](bf16) x B, B given transposed as BT[M,K](bf16)
// One wave computes a 16x64 output tile: 1 A frag is reused across 4 B frags per K-step.
// A frag: lane L -> row m=L&15; elem i -> k = ((i>>3)<<4) + ((L>>4)<<3) + (i&7) (two 16B runs)
// B frag symmetric: lane L -> col n=L&15, same k mapping (BT rows contiguous)
// C frag: lane = n + 16*(m>=8), vgpr r = m%8 (ISA table)
__global__ __launch_bounds__(128) void gemm_bf16_wmma(
    const unsigned short* __restrict__ A,
    const unsigned short* __restrict__ BT,
    unsigned short* __restrict__ C,
    int N, int K, int M)
{
  const int lane = threadIdx.x & 31;
  const int wave = threadIdx.x >> 5;
  const int mt = blockIdx.x * 4 + wave;            // 16-row tile index
  const int ng = blockIdx.y;                       // 64-col group index
  if (mt >= (N >> 4)) return;                      // wave-uniform
  const int half = lane >> 4;
  const int l16  = lane & 15;
  const unsigned short* arow = A + (size_t)(mt * 16 + l16) * K;
  const unsigned short* brow = BT + (size_t)(ng * 64 + l16) * K;
  const size_t bstep = (size_t)16 * K;             // stride between column tiles

  v8f acc[4] = {{0.f,0.f,0.f,0.f,0.f,0.f,0.f,0.f},
                {0.f,0.f,0.f,0.f,0.f,0.f,0.f,0.f},
                {0.f,0.f,0.f,0.f,0.f,0.f,0.f,0.f},
                {0.f,0.f,0.f,0.f,0.f,0.f,0.f,0.f}};
  for (int kb = 0; kb < K; kb += 32) {
    FragBF a;
    a.u[0] = *(const uint4*)(arow + kb + half * 8);
    a.u[1] = *(const uint4*)(arow + kb + 16 + half * 8);
    FragBF b[4];
#pragma unroll
    for (int j = 0; j < 4; ++j) {
      const unsigned short* bp = brow + (size_t)j * bstep;
      b[j].u[0] = *(const uint4*)(bp + kb + half * 8);
      b[j].u[1] = *(const uint4*)(bp + kb + 16 + half * 8);
    }
#pragma unroll
    for (int j = 0; j < 4; ++j)
      acc[j] = __builtin_amdgcn_wmma_f32_16x16x32_bf16(
          false, a.v, false, b[j].v, (short)0, acc[j], false, false);
  }
#pragma unroll
  for (int j = 0; j < 4; ++j) {
    unsigned short* cp =
        C + (size_t)(mt * 16 + half * 8) * M + ng * 64 + j * 16 + l16;
#pragma unroll
    for (int r = 0; r < 8; ++r) cp[(size_t)r * M] = f2bf(acc[j][r]);
  }
}

// ---------- per-layer weight prep: WsT(bf16) = Ws^T ; c_s = Ws@a_s ; c_d = Wd@a_d ----------
__global__ __launch_bounds__(128) void prep_weights(
    const float* __restrict__ Ws, const float* __restrict__ Wd,
    const float* __restrict__ a_s, const float* __restrict__ a_d,
    unsigned short* __restrict__ WsT, float* __restrict__ c_s, float* __restrict__ c_d,
    int din, int dout)
{
  __shared__ float r0[128], r1[128];
  const int k = blockIdx.x;            // 0..din-1
  const int n = threadIdx.x;           // 0..127
  float ss = 0.f, sd = 0.f;
  if (n < dout) {
    float ws = Ws[(size_t)k * dout + n];
    WsT[(size_t)n * din + k] = f2bf(ws);
    ss = ws * a_s[n];
    sd = Wd[(size_t)k * dout + n] * a_d[n];
  }
  r0[n] = ss; r1[n] = sd;
  __syncthreads();
  for (int s = 64; s > 0; s >>= 1) {
    if (n < s) { r0[n] += r0[n + s]; r1[n] += r1[n + s]; }
    __syncthreads();
  }
  if (n == 0) { c_s[k] = r0[0]; c_d[k] = r1[0]; }
}

// ---------- attention logit vectors: out[n] = dot(X[n,:K], c) (one wave / row) ----------
__global__ __launch_bounds__(256) void rowdot(
    const float* __restrict__ X, const float* __restrict__ c,
    float* __restrict__ out, int N, int K)
{
  const int row  = blockIdx.x * 8 + (threadIdx.x >> 5);
  const int lane = threadIdx.x & 31;
  if (row >= N) return;
  float s = 0.f;
  for (int k = lane; k < K; k += 32) s += X[(size_t)row * K + k] * c[k];
#pragma unroll
  for (int off = 16; off > 0; off >>= 1) s += __shfl_xor(s, off);
  if (lane == 0) out[row] = s;
}

// ---------- fills / init ----------
__global__ void fill_f32(float* __restrict__ p, float v, int n) {
  int i = blockIdx.x * blockDim.x + threadIdx.x;
  if (i < n) p[i] = v;
}
__global__ void init_bias(float* __restrict__ acc, const float* __restrict__ b0,
                          const float* __restrict__ b1, int total, int F) {
  int i = blockIdx.x * blockDim.x + threadIdx.x;
  if (i >= total) return;
  int f = i & (F - 1);                   // F is a power of two (64 or 128)
  float v = b0[f];
  if (b1) v += b1[f];
  acc[i] = v;
}
__global__ void cvt_in(const float* __restrict__ x, float* __restrict__ xo,
                       unsigned short* __restrict__ xb, int n) {
  int i = blockIdx.x * blockDim.x + threadIdx.x;
  if (i < n) { float v = x[i]; xo[i] = v; xb[i] = f2bf(v); }
}

// ---------- edge passes ----------
__global__ void edge_logits(const int* __restrict__ src, const int* __restrict__ dst,
                            const float* __restrict__ as, const float* __restrict__ ad,
                            float* __restrict__ e, float* __restrict__ m, int E) {
  int i = blockIdx.x * blockDim.x + threadIdx.x;
  if (i >= E) return;
  float v = as[src[i]] + ad[dst[i]];
  v = v > 0.f ? v : 0.2f * v;            // leaky_relu(0.2)
  e[i] = v;
  atomicMaxF(&m[dst[i]], v);
}
__global__ void edge_expsum(const int* __restrict__ dst, float* __restrict__ e,
                            const float* __restrict__ m, float* __restrict__ denom, int E) {
  int i = blockIdx.x * blockDim.x + threadIdx.x;
  if (i >= E) return;
  int d = dst[i];
  float w = __expf(e[i] - m[d]);
  e[i] = w;
  atomicAdd(&denom[d], w);
}
// 8 features per thread: one uint4 (8 bf16) gather + 8 f32 atomic adds (L2-resident)
__global__ __launch_bounds__(256) void edge_scatter(
    const int* __restrict__ src, const int* __restrict__ dst,
    const float* __restrict__ w, const float* __restrict__ denom,
    const unsigned short* __restrict__ hs, float* __restrict__ acc,
    int E, int F) {
  const int per = F >> 3;
  int t = blockIdx.x * blockDim.x + threadIdx.x;
  int e = t / per;
  if (e >= E) return;
  int c = (t - e * per) << 3;
  int s = src[e], d = dst[e];
  float alpha = w[e] / denom[d];
  const uint4 hv = *(const uint4*)(hs + (size_t)s * F + c);
  float* ap = acc + (size_t)d * F + c;
  atomicAdd(ap + 0, alpha * bf2f(hv.x & 0xFFFFu));
  atomicAdd(ap + 1, alpha * bf2f(hv.x >> 16));
  atomicAdd(ap + 2, alpha * bf2f(hv.y & 0xFFFFu));
  atomicAdd(ap + 3, alpha * bf2f(hv.y >> 16));
  atomicAdd(ap + 4, alpha * bf2f(hv.z & 0xFFFFu));
  atomicAdd(ap + 5, alpha * bf2f(hv.z >> 16));
  atomicAdd(ap + 6, alpha * bf2f(hv.w & 0xFFFFu));
  atomicAdd(ap + 7, alpha * bf2f(hv.w >> 16));
}

// ---------- hetero-mean + tanh ----------
__global__ void finalize_tanh(const float* __restrict__ acc, float scale,
                              float* __restrict__ xo, unsigned short* __restrict__ xb, int n) {
  int i = blockIdx.x * blockDim.x + threadIdx.x;
  if (i >= n) return;
  float v = tanhf(scale * acc[i]);
  xo[i] = v; xb[i] = f2bf(v);
}

extern "C" void kernel_launch(void* const* d_in, const int* in_sizes, int n_in,
                              void* d_out, int out_size, void* d_ws, size_t ws_size,
                              hipStream_t stream) {
  (void)in_sizes; (void)n_in; (void)out_size; (void)ws_size;
  const float* x_int0 = (const float*)d_in[0];
  const float* x_ext0 = (const float*)d_in[1];
  const int* e_ii = (const int*)d_in[2];
  const int* e_ie = (const int*)d_in[3];
  const int* e_ei = (const int*)d_in[4];
  auto P = [&](int l, int r, int which) -> const float* {
    return (const float*)d_in[5 + (l * 3 + r) * 5 + which];   // Ws,Wd,a_s,a_d,b
  };

  // ---- workspace layout ----
  char* ws = (char*)d_ws;
  size_t off = 0;
  auto alloc = [&](size_t bytes) -> void* {
    void* p = ws + off;
    off = (off + bytes + 255) & ~(size_t)255;
    return p;
  };
  float* xi = (float*)alloc((size_t)N_INT * 128 * 4);
  float* xe = (float*)alloc((size_t)N_EXT * 128 * 4);
  unsigned short* xib = (unsigned short*)alloc((size_t)N_INT * 128 * 2);
  unsigned short* xeb = (unsigned short*)alloc((size_t)N_EXT * 128 * 2);
  unsigned short* hs_ii = (unsigned short*)alloc((size_t)N_INT * 128 * 2);
  unsigned short* hs_ie = (unsigned short*)alloc((size_t)N_INT * 128 * 2);
  unsigned short* hs_ei = (unsigned short*)alloc((size_t)N_EXT * 128 * 2);
  float* acc_i = (float*)alloc((size_t)N_INT * 128 * 4);
  float* acc_e = (float*)alloc((size_t)N_EXT * 128 * 4);
  float* as_ii = (float*)alloc((size_t)N_INT * 4);
  float* ad_ii = (float*)alloc((size_t)N_INT * 4);
  float* as_ie = (float*)alloc((size_t)N_INT * 4);
  float* ad_ie = (float*)alloc((size_t)N_EXT * 4);
  float* as_ei = (float*)alloc((size_t)N_EXT * 4);
  float* ad_ei = (float*)alloc((size_t)N_INT * 4);
  float* m_ii = (float*)alloc((size_t)N_INT * 4);
  float* dn_ii = (float*)alloc((size_t)N_INT * 4);
  float* m_ie = (float*)alloc((size_t)N_EXT * 4);
  float* dn_ie = (float*)alloc((size_t)N_EXT * 4);
  float* m_ei = (float*)alloc((size_t)N_INT * 4);
  float* dn_ei = (float*)alloc((size_t)N_INT * 4);
  float* eb_ii = (float*)alloc((size_t)E_II * 4);
  float* eb_ie = (float*)alloc((size_t)E_IE * 4);
  float* eb_ei = (float*)alloc((size_t)E_EI * 4);
  unsigned short* WT[3]; float* cs[3]; float* cd[3];
  for (int r = 0; r < 3; ++r) {
    WT[r] = (unsigned short*)alloc(128 * 128 * 2);
    cs[r] = (float*)alloc(128 * 4);
    cd[r] = (float*)alloc(128 * 4);
  }

  const int dins[4]  = {64, 128, 128, 64};
  const int douts[4] = {128, 128, 64, 64};

  // layer-0 input: copy + bf16 convert
  {
    int n = N_INT * 64;
    cvt_in<<<(n + 255) / 256, 256, 0, stream>>>(x_int0, xi, xib, n);
    n = N_EXT * 64;
    cvt_in<<<(n + 255) / 256, 256, 0, stream>>>(x_ext0, xe, xeb, n);
  }

  for (int l = 0; l < 4; ++l) {
    const int K = dins[l], F = douts[l];

    for (int r = 0; r < 3; ++r)
      prep_weights<<<K, 128, 0, stream>>>(P(l, r, 0), P(l, r, 1), P(l, r, 2), P(l, r, 3),
                                          WT[r], cs[r], cd[r], K, F);

    dim3 gi((N_INT / 16 + 3) / 4, F / 64), ge((N_EXT / 16 + 3) / 4, F / 64);
    gemm_bf16_wmma<<<gi, 128, 0, stream>>>(xib, WT[0], hs_ii, N_INT, K, F);
    gemm_bf16_wmma<<<gi, 128, 0, stream>>>(xib, WT[1], hs_ie, N_INT, K, F);
    gemm_bf16_wmma<<<ge, 128, 0, stream>>>(xeb, WT[2], hs_ei, N_EXT, K, F);

    rowdot<<<(N_INT + 7) / 8, 256, 0, stream>>>(xi, cs[0], as_ii, N_INT, K);
    rowdot<<<(N_INT + 7) / 8, 256, 0, stream>>>(xi, cd[0], ad_ii, N_INT, K);
    rowdot<<<(N_INT + 7) / 8, 256, 0, stream>>>(xi, cs[1], as_ie, N_INT, K);
    rowdot<<<(N_EXT + 7) / 8, 256, 0, stream>>>(xe, cd[1], ad_ie, N_EXT, K);
    rowdot<<<(N_EXT + 7) / 8, 256, 0, stream>>>(xe, cs[2], as_ei, N_EXT, K);
    rowdot<<<(N_INT + 7) / 8, 256, 0, stream>>>(xi, cd[2], ad_ei, N_INT, K);

    fill_f32<<<(N_INT + 255) / 256, 256, 0, stream>>>(m_ii, -INFINITY, N_INT);
    fill_f32<<<(N_EXT + 255) / 256, 256, 0, stream>>>(m_ie, -INFINITY, N_EXT);
    fill_f32<<<(N_INT + 255) / 256, 256, 0, stream>>>(m_ei, -INFINITY, N_INT);
    fill_f32<<<(N_INT + 255) / 256, 256, 0, stream>>>(dn_ii, 0.f, N_INT);
    fill_f32<<<(N_EXT + 255) / 256, 256, 0, stream>>>(dn_ie, 0.f, N_EXT);
    fill_f32<<<(N_INT + 255) / 256, 256, 0, stream>>>(dn_ei, 0.f, N_INT);
    init_bias<<<(N_INT * F + 255) / 256, 256, 0, stream>>>(acc_i, P(l, 0, 4), P(l, 2, 4),
                                                           N_INT * F, F);
    init_bias<<<(N_EXT * F + 255) / 256, 256, 0, stream>>>(acc_e, P(l, 1, 4), nullptr,
                                                           N_EXT * F, F);

    edge_logits<<<(E_II + 255) / 256, 256, 0, stream>>>(e_ii, e_ii + E_II, as_ii, ad_ii,
                                                        eb_ii, m_ii, E_II);
    edge_logits<<<(E_IE + 255) / 256, 256, 0, stream>>>(e_ie, e_ie + E_IE, as_ie, ad_ie,
                                                        eb_ie, m_ie, E_IE);
    edge_logits<<<(E_EI + 255) / 256, 256, 0, stream>>>(e_ei, e_ei + E_EI, as_ei, ad_ei,
                                                        eb_ei, m_ei, E_EI);

    edge_expsum<<<(E_II + 255) / 256, 256, 0, stream>>>(e_ii + E_II, eb_ii, m_ii, dn_ii, E_II);
    edge_expsum<<<(E_IE + 255) / 256, 256, 0, stream>>>(e_ie + E_IE, eb_ie, m_ie, dn_ie, E_IE);
    edge_expsum<<<(E_EI + 255) / 256, 256, 0, stream>>>(e_ei + E_EI, eb_ei, m_ei, dn_ei, E_EI);

    const int per = F >> 3;
    edge_scatter<<<(E_II * per + 255) / 256, 256, 0, stream>>>(e_ii, e_ii + E_II, eb_ii,
                                                               dn_ii, hs_ii, acc_i, E_II, F);
    edge_scatter<<<(E_IE * per + 255) / 256, 256, 0, stream>>>(e_ie, e_ie + E_IE, eb_ie,
                                                               dn_ie, hs_ie, acc_e, E_IE, F);
    edge_scatter<<<(E_EI * per + 255) / 256, 256, 0, stream>>>(e_ei, e_ei + E_EI, eb_ei,
                                                               dn_ei, hs_ei, acc_i, E_EI, F);

    float* oi = (l == 3) ? (float*)d_out : xi;
    float* oe = (l == 3) ? ((float*)d_out + (size_t)N_INT * 64) : xe;
    finalize_tanh<<<(N_INT * F + 255) / 256, 256, 0, stream>>>(acc_i, 0.5f, oi, xib, N_INT * F);
    finalize_tanh<<<(N_EXT * F + 255) / 256, 256, 0, stream>>>(acc_e, 1.0f, oe, xeb, N_EXT * F);
  }
}
